// NonLocalBlock_1297080124235
// MI455X (gfx1250) — matrix operand order repaired and checked by reading, not compiled, not run on hardware
//
#include <hip/hip_runtime.h>
#include <hip/hip_bf16.h>

typedef __bf16 bf16_t;
typedef __attribute__((ext_vector_type(16))) __bf16 v16bf;
typedef __attribute__((ext_vector_type(8)))  float  v8f;
typedef __attribute__((ext_vector_type(2)))  unsigned int u32x2;
typedef int v4i_gcc __attribute__((__vector_size__(16)));

union FragBF {
    v16bf v;
    u32x2 d[4];
    bf16_t h[16];
};

// ---- gfx1250 async global->LDS path (guarded; falls back to VGPR staging) ----
#if defined(__HIP_DEVICE_COMPILE__) &&                                   \
    __has_builtin(__builtin_amdgcn_global_load_async_to_lds_b128) &&     \
    __has_builtin(__builtin_amdgcn_s_wait_asynccnt)
#define HAVE_ASYNC_LDS 1
#else
#define HAVE_ASYNC_LDS 0
#endif

#if HAVE_ASYNC_LDS
__device__ __forceinline__ void async_copy_b128(const void* g, void* l) {
    // Probe-derived signature: (int4* gaddr, int4* lds, imm offset, imm cpol)
    __builtin_amdgcn_global_load_async_to_lds_b128((v4i_gcc*)g, (v4i_gcc*)l, 0, 0);
}
__device__ __forceinline__ void async_wait0() {
    __builtin_amdgcn_s_wait_asynccnt(0);
}
#endif

// ---------------------------------------------------------------- conversions
__global__ void cvt_f32_to_bf16(const float* __restrict__ src,
                                bf16_t* __restrict__ dst, int n) {
    int i = blockIdx.x * blockDim.x + threadIdx.x;
    int stride = gridDim.x * blockDim.x;
    for (; i < n; i += stride) dst[i] = (bf16_t)src[i];
}

// ---------------------------------------------------------------- WMMA GEMM
// C[M,N] = op(A) * op(B), bf16 in, f32 accumulate.
//  TA=false: A is MxK row-major.  TA=true: A stored KxM (lda=M), k-strided.
//  TB=false: B is KxN row-major (k-strided). TB=true: B stored NxK (ldb=K).
// 128x128 block tile, 256 threads (8 waves, 32x64 wave tiles), BK=32,
// double-buffered LDS, one barrier per k-step, async copies when available.
constexpr int BM = 128, BN = 128, BK = 32, LDT = BK + 8;  // 80B rows: 16B aligned

template <bool TA, bool TB, bool OUT_BF16, bool ADD_RES>
__global__ __launch_bounds__(256) void gemm_wmma_bf16(
    const bf16_t* __restrict__ A, const bf16_t* __restrict__ B,
    void* __restrict__ Cout, const float* __restrict__ R,
    int M, int N, int K, int lda, int ldb, int ldc,
    long long sA, long long sB, long long sC, long long sR) {
    __shared__ bf16_t As[2][BM * LDT];
    __shared__ bf16_t Bs[2][BN * LDT];

    const int bz = blockIdx.z;
    A += (long long)bz * sA;
    B += (long long)bz * sB;
    const int n0 = blockIdx.x * BN;
    const int m0 = blockIdx.y * BM;
    (void)M; (void)N;

    const int tid  = threadIdx.x;
    const int lane = tid & 31;
    const int wave = tid >> 5;
    const int wm = (wave >> 1) * 32;  // 4 wave-rows
    const int wn = (wave & 1) * 64;   // 2 wave-cols

    // staging index maps (256 threads)
    const int cr = tid >> 1;          // contiguous: row 0..127
    const int ck = (tid & 1) * 16;    // contiguous: k 0/16
    const int sk = tid >> 3;          // scatter: k 0..31
    const int sc = (tid & 7) * 16;    // scatter: col 0..112

    bf16_t ta[16], tb[16];            // scatter staging temps

    // ---- staging helpers ----
    auto stageA_contig = [&](int k0, bf16_t* dst) {  // !TA: A row-major
        const bf16_t* gp = A + (long long)(m0 + cr) * lda + (k0 + ck);
        bf16_t* sp = dst + cr * LDT + ck;
#if HAVE_ASYNC_LDS
        async_copy_b128(gp, sp);
        async_copy_b128(gp + 8, sp + 8);
#else
        const u32x2* g2 = (const u32x2*)gp;
        *(u32x2*)(sp + 0)  = g2[0];
        *(u32x2*)(sp + 4)  = g2[1];
        *(u32x2*)(sp + 8)  = g2[2];
        *(u32x2*)(sp + 12) = g2[3];
#endif
    };
    auto stageB_contig = [&](int k0, bf16_t* dst) {  // TB: B stored NxK
        const bf16_t* gp = B + (long long)(n0 + cr) * ldb + (k0 + ck);
        bf16_t* sp = dst + cr * LDT + ck;
#if HAVE_ASYNC_LDS
        async_copy_b128(gp, sp);
        async_copy_b128(gp + 8, sp + 8);
#else
        const u32x2* g2 = (const u32x2*)gp;
        *(u32x2*)(sp + 0)  = g2[0];
        *(u32x2*)(sp + 4)  = g2[1];
        *(u32x2*)(sp + 8)  = g2[2];
        *(u32x2*)(sp + 12) = g2[3];
#endif
    };
    auto loadA_scatter = [&](int k0) {  // TA: A stored KxM, contiguous in m
        const bf16_t* gp = A + (long long)(k0 + sk) * lda + (m0 + sc);
        *(u32x2*)&ta[0]  = ((const u32x2*)gp)[0];
        *(u32x2*)&ta[4]  = ((const u32x2*)gp)[1];
        *(u32x2*)&ta[8]  = ((const u32x2*)gp)[2];
        *(u32x2*)&ta[12] = ((const u32x2*)gp)[3];
    };
    auto storeA_scatter = [&](bf16_t* dst) {
#pragma unroll
        for (int i = 0; i < 16; ++i) dst[(sc + i) * LDT + sk] = ta[i];
    };
    auto loadB_scatter = [&](int k0) {  // !TB: B row-major KxN, contiguous in n
        const bf16_t* gp = B + (long long)(k0 + sk) * ldb + (n0 + sc);
        *(u32x2*)&tb[0]  = ((const u32x2*)gp)[0];
        *(u32x2*)&tb[4]  = ((const u32x2*)gp)[1];
        *(u32x2*)&tb[8]  = ((const u32x2*)gp)[2];
        *(u32x2*)&tb[12] = ((const u32x2*)gp)[3];
    };
    auto storeB_scatter = [&](bf16_t* dst) {
#pragma unroll
        for (int i = 0; i < 16; ++i) dst[(sc + i) * LDT + sk] = tb[i];
    };

    v8f acc[2][4] = {};

    // ---- prologue: stage k0=0 into buffer 0 ----
    if constexpr (!TA) stageA_contig(0, As[0]);
    else { loadA_scatter(0); storeA_scatter(As[0]); }
    if constexpr (TB) stageB_contig(0, Bs[0]);
    else { loadB_scatter(0); storeB_scatter(Bs[0]); }
#if HAVE_ASYNC_LDS
    if constexpr (!TA || TB) async_wait0();
#endif
    __syncthreads();

    int pb = 0;
    for (int k0 = 0; k0 < K; k0 += BK) {
        const bool hasNext = (k0 + BK) < K;
        // prefetch next stage (async directly to LDS; scatter ops to VGPRs)
        if (hasNext) {
            if constexpr (!TA) stageA_contig(k0 + BK, As[pb ^ 1]);
            else loadA_scatter(k0 + BK);
            if constexpr (TB) stageB_contig(k0 + BK, Bs[pb ^ 1]);
            else loadB_scatter(k0 + BK);
        }

        // ---- compute from current buffer ----
        // A frag (16x32 bf16): lane<16 -> k {0..7,16..23}, lane>=16 -> +8; m=lane&15
        // B frag (32x16 bf16): lane<16 -> k {0..7,16..23}, lane>=16 -> +8; n=lane&15
        FragBF afr[2], bfr[4];
        const int lrow = lane & 15;
        const int khA = (lane >> 4) << 3;
        const int khB = (lane >> 4) << 3;
        const bf16_t* Ab = As[pb];
        const bf16_t* Bb = Bs[pb];
#pragma unroll
        for (int i = 0; i < 2; ++i) {
            const bf16_t* rp = Ab + (wm + i * 16 + lrow) * LDT + khA;
            afr[i].d[0] = *(const u32x2*)(rp + 0);
            afr[i].d[1] = *(const u32x2*)(rp + 4);
            afr[i].d[2] = *(const u32x2*)(rp + 16);
            afr[i].d[3] = *(const u32x2*)(rp + 20);
        }
#pragma unroll
        for (int i = 0; i < 4; ++i) {
            const bf16_t* rp = Bb + (wn + i * 16 + lrow) * LDT + khB;
            bfr[i].d[0] = *(const u32x2*)(rp + 0);
            bfr[i].d[1] = *(const u32x2*)(rp + 4);
            bfr[i].d[2] = *(const u32x2*)(rp + 16);
            bfr[i].d[3] = *(const u32x2*)(rp + 20);
        }
#pragma unroll
        for (int mi = 0; mi < 2; ++mi)
#pragma unroll
            for (int ni = 0; ni < 4; ++ni)
                acc[mi][ni] = __builtin_amdgcn_wmma_f32_16x16x32_bf16(
                    false, afr[mi].v, false, bfr[ni].v, (short)0, acc[mi][ni],
                    false, false);

        // ---- finish staging next buffer, single barrier ----
        if (hasNext) {
            if constexpr (TA) storeA_scatter(As[pb ^ 1]);
            if constexpr (!TB) storeB_scatter(Bs[pb ^ 1]);
#if HAVE_ASYNC_LDS
            if constexpr (!TA || TB) async_wait0();
#endif
        }
        __syncthreads();
        pb ^= 1;
    }

    // ---- epilogue: C 16x16 f32 frag: m = v + 8*(lane>=16), n = lane&15 ----
    const int lm = (lane >> 4) << 3;
    const int ln = lane & 15;
#pragma unroll
    for (int mi = 0; mi < 2; ++mi) {
#pragma unroll
        for (int ni = 0; ni < 4; ++ni) {
#pragma unroll
            for (int v = 0; v < 8; ++v) {
                const int row = m0 + wm + mi * 16 + lm + v;
                const int col = n0 + wn + ni * 16 + ln;
                const long long idx =
                    (long long)bz * sC + (long long)row * ldc + col;
                if constexpr (OUT_BF16) {
                    ((bf16_t*)Cout)[idx] = (bf16_t)acc[mi][ni][v];
                } else {
                    float val = acc[mi][ni][v];
                    if constexpr (ADD_RES)
                        val += R[(long long)bz * sR + (long long)row * ldc + col];
                    ((float*)Cout)[idx] = val;
                }
            }
        }
    }
}

// ------------------------------------------------------ column softmax (axis n)
// attn is (n x m) row-major; softmax over n for each column m (reference axis=1)
__global__ void softmax_col_kernel(const float* __restrict__ A,
                                   bf16_t* __restrict__ S, int n) {
    const int m = blockIdx.x * blockDim.x + threadIdx.x;
    if (m >= n) return;
    float mx = -INFINITY;
    for (int i = 0; i < n; ++i) mx = fmaxf(mx, A[(long long)i * n + m]);
    float sum = 0.0f;
    for (int i = 0; i < n; ++i) sum += __expf(A[(long long)i * n + m] - mx);
    const float inv = 1.0f / sum;
    for (int i = 0; i < n; ++i)
        S[(long long)i * n + m] =
            (bf16_t)(__expf(A[(long long)i * n + m] - mx) * inv);
}

// ---------------------------------------------------------------- launcher
extern "C" void kernel_launch(void* const* d_in, const int* in_sizes, int n_in,
                              void* d_out, int out_size, void* d_ws,
                              size_t ws_size, hipStream_t stream) {
    const float* x       = (const float*)d_in[0];
    const float* w_phi   = (const float*)d_in[1];
    const float* w_theta = (const float*)d_in[2];
    const float* w_g     = (const float*)d_in[3];
    const float* w_mask  = (const float*)d_in[4];
    float* out = (float*)d_out;

    constexpr int Bt = 8, C = 256, IC = 128, HW = 4096, Nn = 2048;

    char* p = (char*)d_ws;
    auto carve = [&](size_t bytes) {
        char* q = p;
        p += (bytes + 255) & ~(size_t)255;
        return q;
    };
    bf16_t* xb      = (bf16_t*)carve((size_t)Bt * C * HW * 2);
    bf16_t* wphib   = (bf16_t*)carve((size_t)IC * C * 2);
    bf16_t* wthetab = (bf16_t*)carve((size_t)IC * C * 2);
    bf16_t* wgb     = (bf16_t*)carve((size_t)IC * C * 2);
    bf16_t* wmaskb  = (bf16_t*)carve((size_t)C * IC * 2);
    bf16_t* phi     = (bf16_t*)carve((size_t)Bt * IC * HW * 2);
    bf16_t* theta   = (bf16_t*)carve((size_t)Bt * IC * HW * 2);
    bf16_t* g       = (bf16_t*)carve((size_t)Bt * IC * HW * 2);
    bf16_t* yb      = (bf16_t*)carve((size_t)Bt * IC * HW * 2);
    bf16_t* smx     = (bf16_t*)carve((size_t)Bt * Nn * Nn * 2);  // all batches
    float*  attn    = (float*)carve((size_t)Nn * Nn * 4);        // per-batch reuse
    (void)ws_size; (void)in_sizes; (void)n_in; (void)out_size;

    // bf16 conversions
    {
        int n = Bt * C * HW;
        cvt_f32_to_bf16<<<(n + 255) / 256, 256, 0, stream>>>(x, xb, n);
        cvt_f32_to_bf16<<<(IC * C + 255) / 256, 256, 0, stream>>>(w_phi, wphib, IC * C);
        cvt_f32_to_bf16<<<(IC * C + 255) / 256, 256, 0, stream>>>(w_theta, wthetab, IC * C);
        cvt_f32_to_bf16<<<(IC * C + 255) / 256, 256, 0, stream>>>(w_g, wgb, IC * C);
        cvt_f32_to_bf16<<<(C * IC + 255) / 256, 256, 0, stream>>>(w_mask, wmaskb, C * IC);
    }

    const dim3 blk(256);

    // Projections (batched over z): C(128x4096) = W(128x256) x Xb(256x4096)
    {
        dim3 grd(HW / BN, IC / BM, Bt);
        gemm_wmma_bf16<false, false, true, false><<<grd, blk, 0, stream>>>(
            wphib, xb, phi, nullptr, IC, HW, C, C, HW, HW,
            0, (long long)C * HW, (long long)IC * HW, 0);
        gemm_wmma_bf16<false, false, true, false><<<grd, blk, 0, stream>>>(
            wthetab, xb, theta, nullptr, IC, HW, C, C, HW, HW,
            0, (long long)C * HW, (long long)IC * HW, 0);
        gemm_wmma_bf16<false, false, true, false><<<grd, blk, 0, stream>>>(
            wgb, xb, g, nullptr, IC, HW, C, C, HW, HW,
            0, (long long)C * HW, (long long)IC * HW, 0);
    }

    // Attention logits + softmax per batch (attn scratch reused).
    // phi/theta/g (128x4096) reinterpret as (256x2048) row-major for free.
    for (int b = 0; b < Bt; ++b) {
        const bf16_t* Theta2 = theta + (long long)b * IC * HW;  // (256 x 2048)
        const bf16_t* Phi2   = phi   + (long long)b * IC * HW;  // (256 x 2048)

        // attn(n,m) = sum_c Theta2[c,n] * Phi2[c,m]  -> TRANS_A GEMM, f32 out
        gemm_wmma_bf16<true, false, false, false>
            <<<dim3(Nn / BN, Nn / BM, 1), blk, 0, stream>>>(
                Theta2, Phi2, attn, nullptr, Nn, Nn, C, Nn, Nn, Nn, 0, 0, 0, 0);

        // softmax over n (axis=1 of (b,n,m)), emit bf16 S
        softmax_col_kernel<<<Nn / 256, 256, 0, stream>>>(
            attn, smx + (long long)b * Nn * Nn, Nn);
    }

    // Y2(c,n) = sum_m G2[c,m] * S[n,m]  -> TRANS_B GEMM, bf16 out (batched z)
    gemm_wmma_bf16<false, true, true, false>
        <<<dim3(Nn / BN, C / BM, Bt), blk, 0, stream>>>(
            g, smx, yb, nullptr, C, Nn, Nn, Nn, Nn, Nn,
            (long long)IC * HW, (long long)Nn * Nn, (long long)IC * HW, 0);

    // mask + residual (batched): out(256x4096) = Wm(256x128) x Y(128x4096) + x
    {
        dim3 grd(HW / BN, C / BM, Bt);
        gemm_wmma_bf16<false, false, false, true><<<grd, blk, 0, stream>>>(
            wmaskb, yb, out, x, C, HW, IC, IC, HW, HW,
            0, (long long)IC * HW, (long long)C * HW, (long long)C * HW);
    }
}
